// MaskedRNN_26001732010257
// MI455X (gfx1250) — matrix operand order repaired and checked by reading, not compiled, not run on hardware
//
#include <hip/hip_runtime.h>
#include <hip/hip_bf16.h>

typedef __attribute__((ext_vector_type(16))) __bf16 v16bf;
typedef __attribute__((ext_vector_type(8)))  __bf16 v8bf;
typedef __attribute__((ext_vector_type(8)))  float  v8f;

#define B_  64
#define S_  512
#define I_  1024
#define H_  1024

// ---------------- utility kernels ----------------

__global__ void cvt_f32_bf16(const float* __restrict__ in, __bf16* __restrict__ out, int n) {
    int i = blockIdx.x * blockDim.x + threadIdx.x;
    if (i < n) out[i] = (__bf16)in[i];
}

__global__ void zero_bf16(__bf16* __restrict__ p, int n) {
    int i = blockIdx.x * blockDim.x + threadIdx.x;
    if (i < n) p[i] = (__bf16)0.0f;
}

// ---------------- WMMA fragment loaders (ISA 7.12.2 layouts) ----------------

// A: 16x32 bf16 tile, row-major source with leading dim ld.
// lanes 0-15 : M=lane,    elements = K 0..7  then 16..23
// lanes 16-31: M=lane-16, elements = K 8..15 then 24..31
__device__ __forceinline__ v16bf load_a_frag(const __bf16* __restrict__ base,
                                             int lane, int row_tile, int k0, int ld) {
    int row   = row_tile + (lane & 15);
    int kbase = (lane >> 4) << 3;                 // 0 or 8
    const __bf16* p = base + (size_t)row * ld + k0 + kbase;
    v8bf lo = *reinterpret_cast<const v8bf*>(p);        // K = kbase .. kbase+7
    v8bf hi = *reinterpret_cast<const v8bf*>(p + 16);   // K = 16+kbase .. 16+kbase+7
    v16bf a;
#pragma unroll
    for (int j = 0; j < 8; ++j) { a[j] = lo[j]; a[j + 8] = hi[j]; }
    return a;
}

// B: 32x16 bf16 tile where B[k][n] = Wt[n][k] and Wt is row-major [N][K] with ld.
// lanes 0-15 : N=lane,    elements = K 0..7  then 16..23
// lanes 16-31: N=lane-16, elements = K 8..15 then 24..31
__device__ __forceinline__ v16bf load_b_frag(const __bf16* __restrict__ base,
                                             int lane, int col_tile, int k0, int ld) {
    int n  = col_tile + (lane & 15);
    int kb = (lane >> 4) << 3;                    // 0 or 8
    const __bf16* p = base + (size_t)n * ld + k0 + kb;
    v8bf lo = *reinterpret_cast<const v8bf*>(p);
    v8bf hi = *reinterpret_cast<const v8bf*>(p + 16);
    v16bf b;
#pragma unroll
    for (int j = 0; j < 8; ++j) { b[j] = lo[j]; b[j + 8] = hi[j]; }
    return b;
}

// ---------------- phase 1: all-timestep input GEMMs + mask fusion ----------------
// wx[s][b][h] = sigmoid(x W_mask^T) * (x W^T), X flattened [B*S, I] (row = b*S + s)

__global__ void __launch_bounds__(256)
phase1_xgemm(const __bf16* __restrict__ x16,   // [B*S, I]
             const __bf16* __restrict__ W16,   // [H, I]
             const __bf16* __restrict__ Wm16,  // [H, I]
             float* __restrict__ wx)           // [S, B, H]
{
    int lane = threadIdx.x & 31;
    int job  = (int)((blockIdx.x * blockDim.x + threadIdx.x) >> 5);
    int mt   = job >> 6;          // 0..2047  (32768/16 m-tiles)
    int nt   = job & 63;          // 0..63    (1024/16 n-tiles)
    int mrow = mt << 4;
    int ncol = nt << 4;

    v8f acc_w = {};
    v8f acc_m = {};
    for (int k0 = 0; k0 < I_; k0 += 32) {
        v16bf a  = load_a_frag(x16,  lane, mrow, k0, I_);
        v16bf bw = load_b_frag(W16,  lane, ncol, k0, I_);
        v16bf bm = load_b_frag(Wm16, lane, ncol, k0, I_);
        acc_w = __builtin_amdgcn_wmma_f32_16x16x32_bf16(false, a, false, bw,
                                                        (short)0, acc_w, false, false);
        acc_m = __builtin_amdgcn_wmma_f32_16x16x32_bf16(false, a, false, bm,
                                                        (short)0, acc_m, false, false);
    }

    int n     = ncol + (lane & 15);
    int mbase = mrow + ((lane >> 4) << 3);
#pragma unroll
    for (int r = 0; r < 8; ++r) {
        int m = mbase + r;                 // row in [0, B*S)
        int b = m >> 9;                    // / S_
        int s = m & (S_ - 1);
        float g = 1.0f / (1.0f + __expf(-acc_m[r]));
        wx[(size_t)s * (B_ * H_) + (size_t)b * H_ + n] = g * acc_w[r];
    }
}

// ---------------- phase 2: one recurrent step ----------------
// h_new = tanh( wx_s + sigmoid(h U_mask^T) * (h U^T) + bias )

__global__ void __launch_bounds__(256)
step_kernel(const __bf16* __restrict__ h_in,   // [B, H] bf16
            __bf16*       __restrict__ h_out,  // [B, H] bf16
            const __bf16* __restrict__ U16,    // [H, H]
            const __bf16* __restrict__ Um16,   // [H, H]
            const float*  __restrict__ wx_s,   // [B, H] fp32 (slice for step s)
            const float*  __restrict__ bias,   // [H]
            float*        __restrict__ out,    // [B, S, H]
            int s)
{
    int lane = threadIdx.x & 31;
    int job  = (int)((blockIdx.x * blockDim.x + threadIdx.x) >> 5);  // 0..255
    int mt   = job >> 6;          // 0..3   (64/16 m-tiles)
    int nt   = job & 63;          // 0..63
    int mrow = mt << 4;
    int ncol = nt << 4;

    v8f acc_u = {};
    v8f acc_m = {};
    for (int k0 = 0; k0 < H_; k0 += 32) {
        v16bf a  = load_a_frag(h_in, lane, mrow, k0, H_);
        v16bf bu = load_b_frag(U16,  lane, ncol, k0, H_);
        v16bf bm = load_b_frag(Um16, lane, ncol, k0, H_);
        acc_u = __builtin_amdgcn_wmma_f32_16x16x32_bf16(false, a, false, bu,
                                                        (short)0, acc_u, false, false);
        acc_m = __builtin_amdgcn_wmma_f32_16x16x32_bf16(false, a, false, bm,
                                                        (short)0, acc_m, false, false);
    }

    int n     = ncol + (lane & 15);
    int mbase = mrow + ((lane >> 4) << 3);
    float bn  = bias[n];
#pragma unroll
    for (int r = 0; r < 8; ++r) {
        int m = mbase + r;                 // batch index
        float g  = 1.0f / (1.0f + __expf(-acc_m[r]));
        float v  = g * acc_u[r] + wx_s[(size_t)m * H_ + n] + bn;
        float hv = tanhf(v);
        out[(size_t)m * (S_ * H_) + (size_t)s * H_ + n] = hv;
        h_out[(size_t)m * H_ + n] = (__bf16)hv;
    }
}

// ---------------- host launcher ----------------

extern "C" void kernel_launch(void* const* d_in, const int* in_sizes, int n_in,
                              void* d_out, int out_size, void* d_ws, size_t ws_size,
                              hipStream_t stream) {
    const float* x    = (const float*)d_in[0];
    const float* W    = (const float*)d_in[1];
    const float* Wm   = (const float*)d_in[2];
    const float* U    = (const float*)d_in[3];
    const float* Um   = (const float*)d_in[4];
    const float* bias = (const float*)d_in[5];
    float* out = (float*)d_out;

    // workspace carve-out (all 256B-aligned sizes)
    char* ws = (char*)d_ws;
    size_t off = 0;
    __bf16* x16  = (__bf16*)(ws + off); off += (size_t)B_ * S_ * I_ * 2;  // 64 MB
    __bf16* W16  = (__bf16*)(ws + off); off += (size_t)H_ * I_ * 2;       // 2 MB
    __bf16* Wm16 = (__bf16*)(ws + off); off += (size_t)H_ * I_ * 2;
    __bf16* U16  = (__bf16*)(ws + off); off += (size_t)H_ * H_ * 2;
    __bf16* Um16 = (__bf16*)(ws + off); off += (size_t)H_ * H_ * 2;
    float*  wx   = (float*)(ws + off);  off += (size_t)S_ * B_ * H_ * 4;  // 128 MB
    __bf16* h0   = (__bf16*)(ws + off); off += (size_t)B_ * H_ * 2;
    __bf16* h1   = (__bf16*)(ws + off); off += (size_t)B_ * H_ * 2;

    // bf16 conversions of x and all four weight matrices
    int nx = B_ * S_ * I_;
    cvt_f32_bf16<<<(nx + 255) / 256, 256, 0, stream>>>(x, x16, nx);
    int nw = H_ * I_;
    cvt_f32_bf16<<<(nw + 255) / 256, 256, 0, stream>>>(W,  W16,  nw);
    cvt_f32_bf16<<<(nw + 255) / 256, 256, 0, stream>>>(Wm, Wm16, nw);
    cvt_f32_bf16<<<(nw + 255) / 256, 256, 0, stream>>>(U,  U16,  nw);
    cvt_f32_bf16<<<(nw + 255) / 256, 256, 0, stream>>>(Um, Um16, nw);

    int nh = B_ * H_;
    zero_bf16<<<(nh + 255) / 256, 256, 0, stream>>>(h0, nh);

    // phase 1: 2048 m-tiles x 64 n-tiles, one wave per tile, 8 waves per block
    phase1_xgemm<<<(2048 * 64) / 8, 256, 0, stream>>>(x16, W16, Wm16, wx);

    // phase 2: 512 dependent steps, ping-pong hidden state in bf16
    __bf16* hbuf[2] = {h0, h1};
    for (int s = 0; s < S_; ++s) {
        step_kernel<<<(4 * 64) / 8, 256, 0, stream>>>(
            hbuf[s & 1], hbuf[(s + 1) & 1], U16, Um16,
            wx + (size_t)s * B_ * H_, bias, out, s);
    }

    (void)in_sizes; (void)n_in; (void)out_size; (void)ws_size;
}
